// Location_enhancement_Multimodal_alignment_27728308863474
// MI455X (gfx1250) — matrix operand order repaired and checked by reading, not compiled, not run on hardware
//
#include <hip/hip_runtime.h>
#include <hip/hip_bf16.h>

// MI455X (gfx1250, wave32) implementation.
// All dense GEMMs (QKV/O/FFN projections, attn_map x PE einsum, 3x3 convs as
// implicit GEMM) run on v_wmma_f32_16x16x32_bf16 (bf16 operands, f32 accum).
// Every GEMM here has Out == 64, so one wave computes a 16x64 strip: the A
// fragment is loaded/converted once per K-chunk and reused by 4 back-to-back
// WMMAs into 4 resident f32 accumulators (4x less activation traffic, 4x WMMA
// issue density). Attention cores (dh=16), softmax, LayerNorm, instance-norm,
// bitonic argsort, gathers are VALU/LDS kernels.

typedef __attribute__((ext_vector_type(16))) __bf16 v16bf;
typedef __attribute__((ext_vector_type(8)))  float  v8f;

#define kN   256
#define kC   64
#define kT   26
#define kHW  1024
#define kD   64
#define kSEL 500
#define kNH  4
#define kCLS 37
#define EPSF 1e-5f
#define CEILD(a,b) (((a)+(b)-1)/(b))

static __device__ __forceinline__ float wred_sum(float v) {
#pragma unroll
  for (int o = 16; o > 0; o >>= 1) v += __shfl_xor(v, o, 32);
  return v;
}
static __device__ __forceinline__ float wred_max(float v) {
#pragma unroll
  for (int o = 16; o > 0; o >>= 1) v = fmaxf(v, __shfl_xor(v, o, 32));
  return v;
}

// ---------------------------------------------------------------------------
// WMMA GEMM (Out == 64): Y[M,64] = A[M,K] * W[64,K]^T (+bias) (+relu).
// K % 32 == 0, M % 16 == 0. One wave computes a 16x64 strip (4 N-tiles).
// A-frag (16x32 bf16): lane m=lane&15, half=lane>>4; e<8 -> k=half*8+e,
// e>=8 -> k=16+half*8+(e-8).  B-frag (32x16): b[e]=W[n0+(lane&15)][k0+k(e)], same k order as A.
// C/D: c[r] = Y[m0 + r + 8*half][n0 + (lane&15)].
// ---------------------------------------------------------------------------
__global__ void k_gemm_xwt64(const float* __restrict__ A, const float* __restrict__ W,
                             const float* __restrict__ bias, float* __restrict__ Y,
                             int M, int K, int relu) {
  int tid  = blockIdx.x * blockDim.x + threadIdx.x;
  int wave = tid >> 5;
  int lane = threadIdx.x & 31;
  if (wave >= (M >> 4)) return;
  int m0 = wave << 4;
  int l15 = lane & 15, hf = lane >> 4;
  const float* arow = A + (size_t)(m0 + l15) * K;
  v8f acc0 = {0.f,0.f,0.f,0.f,0.f,0.f,0.f,0.f};
  v8f acc1 = acc0, acc2 = acc0, acc3 = acc0;
  for (int k0 = 0; k0 < K; k0 += 32) {
    v16bf a;
    const float* ap = arow + k0 + hf * 8;
    __builtin_prefetch(ap + 32, 0, 3); // next K-chunk of this row
#pragma unroll
    for (int e = 0; e < 8; ++e) a[e] = (__bf16)ap[e];
#pragma unroll
    for (int e = 0; e < 8; ++e) a[8 + e] = (__bf16)ap[16 + e];
#pragma unroll
    for (int nt = 0; nt < 4; ++nt) {
      v16bf b;
      const float* bp = W + (size_t)(nt * 16 + l15) * K + k0 + hf * 8;
#pragma unroll
      for (int e = 0; e < 8; ++e) { b[e] = (__bf16)bp[e]; b[8 + e] = (__bf16)bp[16 + e]; }
      v8f* accp = (nt == 0) ? &acc0 : (nt == 1) ? &acc1 : (nt == 2) ? &acc2 : &acc3;
      *accp = __builtin_amdgcn_wmma_f32_16x16x32_bf16(false, a, false, b, (short)0, *accp, false, false);
    }
  }
#pragma unroll
  for (int nt = 0; nt < 4; ++nt) {
    v8f acc = (nt == 0) ? acc0 : (nt == 1) ? acc1 : (nt == 2) ? acc2 : acc3;
    int n = nt * 16 + l15;
    float bv = bias ? bias[n] : 0.f;
#pragma unroll
    for (int r = 0; r < 8; ++r) {
      float v = acc[r] + bv;
      if (relu) v = fmaxf(v, 0.f);
      Y[((size_t)(m0 + r + 8 * hf) << 6) + n] = v;
    }
  }
}

// ---------------------------------------------------------------------------
// Implicit-GEMM 3x3 SAME conv via WMMA (Co == 64). X:(N,Ci,32,32),
// Wf:(64,Ci,3,3). M = N*1024, K = Ci*9 (288 or 576, both %32==0).
// One wave computes a 16x64 strip: the gathered/bounds-checked A fragment is
// built once per K-chunk and reused by 4 WMMAs.
// layout 0: Y:(N,64,1024). layout 1 (dwc): Y[hw][n][co] = acc + bias + pe[hw][co].
// ---------------------------------------------------------------------------
__global__ void k_conv3x3_wmma64(const float* __restrict__ X, const float* __restrict__ Wf,
                                 const float* __restrict__ bias, const float* __restrict__ pe,
                                 float* __restrict__ Y, int Ci, int layout) {
  int tid  = blockIdx.x * blockDim.x + threadIdx.x;
  int wave = tid >> 5;
  int lane = threadIdx.x & 31;
  int M = kN * kHW;
  if (wave >= (M >> 4)) return;
  int m0 = wave << 4;
  int l15 = lane & 15, hf = lane >> 4;
  int K = Ci * 9;
  int m = m0 + l15;
  int nimg = m >> 10; int hw = m & 1023; int hy = hw >> 5; int wx = hw & 31;
  const float* xb = X + ((size_t)nimg * Ci << 10);
  v8f acc0 = {0.f,0.f,0.f,0.f,0.f,0.f,0.f,0.f};
  v8f acc1 = acc0, acc2 = acc0, acc3 = acc0;
  for (int k0 = 0; k0 < K; k0 += 32) {
    v16bf a;
#pragma unroll
    for (int e = 0; e < 16; ++e) {
      int k  = k0 + ((e < 8) ? 0 : 16) + hf * 8 + (e & 7);
      int ci = k / 9, r = k - ci * 9;
      int yy = hy + r / 3 - 1, xx = wx + (r % 3) - 1;
      float vv = 0.f;
      if ((unsigned)yy < 32u && (unsigned)xx < 32u)
        vv = xb[((size_t)ci << 10) + (yy << 5) + xx];
      a[e] = (__bf16)vv;
    }
#pragma unroll
    for (int nt = 0; nt < 4; ++nt) {
      v16bf b;
      const float* bp = Wf + (size_t)(nt * 16 + l15) * K + k0 + hf * 8;
#pragma unroll
      for (int e = 0; e < 8; ++e) { b[e] = (__bf16)bp[e]; b[8 + e] = (__bf16)bp[16 + e]; }
      v8f* accp = (nt == 0) ? &acc0 : (nt == 1) ? &acc1 : (nt == 2) ? &acc2 : &acc3;
      *accp = __builtin_amdgcn_wmma_f32_16x16x32_bf16(false, a, false, b, (short)0, *accp, false, false);
    }
  }
#pragma unroll
  for (int nt = 0; nt < 4; ++nt) {
    v8f acc = (nt == 0) ? acc0 : (nt == 1) ? acc1 : (nt == 2) ? acc2 : acc3;
    int nn = nt * 16 + l15;
    float bv = bias ? bias[nn] : 0.f;
#pragma unroll
    for (int r = 0; r < 8; ++r) {
      int mm = m0 + r + 8 * hf;
      int nb = mm >> 10, hw2 = mm & 1023;
      float v = acc[r] + bv;
      if (layout == 0) {
        Y[(((size_t)nb << 6) + nn << 10) + hw2] = v;
      } else { // (HW, N, 64) with sinusoid PE added
        Y[(((size_t)hw2 * kN + nb) << 6) + nn] = v + pe[(hw2 << 6) + nn];
      }
    }
  }
}

// ------------------------- sinusoid PE (and transpose) ---------------------
__global__ void k_init_pe(float* __restrict__ PE, float* __restrict__ PET) {
  int id = blockIdx.x * blockDim.x + threadIdx.x;
  if (id >= 1024 * 32) return;
  int pos = id >> 5; int i = id & 31;
  float freq = expf((float)(2 * i) * (-9.210340371976184f / 64.f));
  float ang = (float)pos * freq;
  float s = sinf(ang), c = cosf(ang);
  PE[(pos << 6) + 2 * i]     = s;
  PE[(pos << 6) + 2 * i + 1] = c;
  PET[((size_t)(2 * i) << 10) + pos]     = s;
  PET[((size_t)(2 * i + 1) << 10) + pos] = c;
}

// ------------------- pos_weight = max_t attn*mask ---------------------------
__global__ void k_posweight(const float* __restrict__ AM, const int* __restrict__ len,
                            float* __restrict__ PW) {
  int id = blockIdx.x * blockDim.x + threadIdx.x;
  if (id >= kN * kHW) return;
  int n = id >> 10; int hw = id & 1023;
  int L = len[n];
  float m = -INFINITY;
  for (int t = 0; t < kT; ++t) {
    float v = AM[(((size_t)n * kT + t) << 10) + hw];
    m = fmaxf(m, (t < L) ? v : 0.f);
  }
  PW[id] = m;
}

// ----------------- conv1 (1->32) direct + BN + PReLU ------------------------
__global__ void k_conv1_bn_prelu(const float* __restrict__ PW, const float* __restrict__ Wf,
                                 const float* __restrict__ g, const float* __restrict__ b,
                                 const float* __restrict__ m, const float* __restrict__ v,
                                 const float* __restrict__ al, float* __restrict__ Y) {
  int id = blockIdx.x * blockDim.x + threadIdx.x;
  if (id >= kN * 32 * kHW) return;
  int hw = id & 1023; int c = (id >> 10) & 31; int n = id >> 15;
  int hy = hw >> 5, wx = hw & 31;
  float acc = 0.f;
#pragma unroll
  for (int dy = 0; dy < 3; ++dy)
#pragma unroll
    for (int dx = 0; dx < 3; ++dx) {
      int yy = hy + dy - 1, xx = wx + dx - 1;
      if ((unsigned)yy < 32u && (unsigned)xx < 32u)
        acc += PW[((size_t)n << 10) + (yy << 5) + xx] * Wf[c * 9 + dy * 3 + dx];
    }
  float sc = g[c] * rsqrtf(v[c] + EPSF);
  float y = (acc - m[c]) * sc + b[c];
  float a = al[0];
  Y[id] = (y >= 0.f) ? y : a * y;
}

// --------------------------- BN + PReLU (in place) --------------------------
__global__ void k_bn_prelu(float* __restrict__ X, const float* __restrict__ g,
                           const float* __restrict__ b, const float* __restrict__ m,
                           const float* __restrict__ v, const float* __restrict__ al,
                           int total, int Cc) {
  int id = blockIdx.x * blockDim.x + threadIdx.x;
  if (id >= total) return;
  int c = (id >> 10) % Cc;
  float sc = g[c] * rsqrtf(v[c] + EPSF);
  float y = (X[id] - m[c]) * sc + b[c];
  float a = al[0];
  X[id] = (y >= 0.f) ? y : a * y;
}

// --------------------- row softmax over 1024 (in place) ---------------------
__global__ void k_softmax1024(float* __restrict__ X, int rows) {
  int wid  = (blockIdx.x * blockDim.x + threadIdx.x) >> 5;
  int lane = threadIdx.x & 31;
  if (wid >= rows) return;
  float* row = X + ((size_t)wid << 10);
  float v[32]; float mx = -INFINITY;
#pragma unroll
  for (int i = 0; i < 32; ++i) { v[i] = row[(i << 5) + lane]; mx = fmaxf(mx, v[i]); }
  mx = wred_max(mx);
  float s = 0.f;
#pragma unroll
  for (int i = 0; i < 32; ++i) { v[i] = expf(v[i] - mx); s += v[i]; }
  s = wred_sum(s);
  float inv = 1.f / s;
#pragma unroll
  for (int i = 0; i < 32; ++i) row[(i << 5) + lane] = v[i] * inv;
}

// -------------- instance-norm over HW, multiplied by pw1 --------------------
__global__ void k_innorm_mul(const float* __restrict__ X, const float* __restrict__ PW1,
                             float* __restrict__ Y, int rows) {
  int wid  = (blockIdx.x * blockDim.x + threadIdx.x) >> 5;
  int lane = threadIdx.x & 31;
  if (wid >= rows) return;
  size_t base = (size_t)wid << 10;
  float v[32]; float s = 0.f;
#pragma unroll
  for (int i = 0; i < 32; ++i) { v[i] = X[base + (i << 5) + lane]; s += v[i]; }
  float mu = wred_sum(s) * (1.f / 1024.f);
  float q = 0.f;
#pragma unroll
  for (int i = 0; i < 32; ++i) { float d = v[i] - mu; q += d * d; }
  float var = wred_sum(q) * (1.f / 1024.f);
  float rs = rsqrtf(var + EPSF);
#pragma unroll
  for (int i = 0; i < 32; ++i) {
    size_t idx = base + (i << 5) + lane;
    Y[idx] = (v[i] - mu) * rs * PW1[idx];
  }
}

// ------------- per-batch descending argsort of 1024 (bitonic, LDS) ----------
__global__ void k_argsort1024(const float* __restrict__ PW, int* __restrict__ order) {
  __shared__ unsigned long long keys[1024];
  int n = blockIdx.x, t = threadIdx.x;
  float f = PW[((size_t)n << 10) + t];
  unsigned u = __float_as_uint(f);
  u = (u & 0x80000000u) ? ~u : (u | 0x80000000u); // monotone ascending map
  u = ~u;                                          // descending on value
  keys[t] = ((unsigned long long)u << 32) | (unsigned)t; // tie-break: idx asc
  __syncthreads();
  for (int k = 2; k <= 1024; k <<= 1) {
    for (int j = k >> 1; j > 0; j >>= 1) {
      int ixj = t ^ j;
      unsigned long long a = keys[t];
      unsigned long long b = keys[ixj];
      bool dir = ((t & k) == 0);
      bool dosw = (ixj > t) && ((a > b) == dir);
      __syncthreads();
      if (dosw) { keys[t] = b; keys[ixj] = a; }
      __syncthreads();
    }
  }
  order[((size_t)n << 10) + t] = (int)(keys[t] & 0xffffffffull);
}

// ---------------------- gather / zero by rank -------------------------------
__global__ void k_gather_select(const float* __restrict__ PEF_, const int* __restrict__ order,
                                float* __restrict__ S) {
  int id = blockIdx.x * blockDim.x + threadIdx.x;
  if (id >= kSEL * kN * kD) return;
  int c = id & 63; int n = (id >> 6) & 255; int s = id >> 14;
  int row = order[((size_t)n << 10) + s];
  S[id] = PEF_[(((size_t)row * kN + n) << 6) + c];
}
__global__ void k_zero_unsel(float* __restrict__ PEF_, const int* __restrict__ order) {
  int id = blockIdx.x * blockDim.x + threadIdx.x;
  if (id >= (kHW - kSEL) * kN * kD) return;
  int c = id & 63; int n = (id >> 6) & 255; int r = id >> 14;
  int row = order[((size_t)n << 10) + kSEL + r];
  PEF_[(((size_t)row * kN + n) << 6) + c] = 0.f;
}

// --------- tf = text_logits @ Wp^T (K=37, VALU) + einsum part ---------------
__global__ void k_textproj(const float* __restrict__ TL, const float* __restrict__ Wp,
                           const float* __restrict__ PEADD, float* __restrict__ TF) {
  int id = blockIdx.x * blockDim.x + threadIdx.x;
  if (id >= kT * kN * kD) return;
  int d = id & 63; int n = (id >> 6) & 255; int t = id >> 14;
  const float* x = TL + ((size_t)n * kT + t) * kCLS;
  const float* w = Wp + (size_t)d * kCLS;
  float s = 0.f;
  for (int j = 0; j < kCLS; ++j) s += x[j] * w[j];
  TF[id] = s + PEADD[(((size_t)n * kT + t) << 6) + d];
}

__global__ void k_copy(const float* __restrict__ A, float* __restrict__ B, int n) {
  int id = blockIdx.x * blockDim.x + threadIdx.x;
  if (id < n) B[id] = A[id];
}

// --------------------- MHA core: wave per (q, n, h), dh=16 ------------------
__global__ void k_mha(const float* __restrict__ Q, const float* __restrict__ Kb,
                      const float* __restrict__ Vb, float* __restrict__ O,
                      int Qs, int Ks, int Nb) {
  int wid  = (blockIdx.x * blockDim.x + threadIdx.x) >> 5;
  int lane = threadIdx.x & 31;
  int total = Qs * Nb * kNH;
  if (wid >= total) return;
  int h = wid & 3;
  int n = (wid >> 2) % Nb;
  int q = wid / (Nb * kNH);
  const float* qp = Q + (((size_t)q * Nb + n) << 6) + h * 16;
  float qv[16];
#pragma unroll
  for (int d = 0; d < 16; ++d) qv[d] = qp[d] * 0.25f; // dh^-0.5
  float sc[16]; float mx = -INFINITY;
#pragma unroll
  for (int i = 0; i < 16; ++i) { // supports Ks <= 512
    int k = lane + (i << 5);
    float s = -INFINITY;
    if (k < Ks) {
      const float* kp = Kb + (((size_t)k * Nb + n) << 6) + h * 16;
      s = 0.f;
#pragma unroll
      for (int d = 0; d < 16; ++d) s += qv[d] * kp[d];
    }
    sc[i] = s; mx = fmaxf(mx, s);
  }
  mx = wred_max(mx);
  float sum = 0.f;
#pragma unroll
  for (int i = 0; i < 16; ++i) {
    float e = (sc[i] == -INFINITY) ? 0.f : expf(sc[i] - mx);
    sc[i] = e; sum += e;
  }
  sum = wred_sum(sum);
  float inv = 1.f / sum;
  float acc[16];
#pragma unroll
  for (int d = 0; d < 16; ++d) acc[d] = 0.f;
#pragma unroll
  for (int i = 0; i < 16; ++i) {
    int k = lane + (i << 5);
    if (k < Ks) {
      const float* vp = Vb + (((size_t)k * Nb + n) << 6) + h * 16;
      float w = sc[i];
#pragma unroll
      for (int d = 0; d < 16; ++d) acc[d] += w * vp[d];
    }
  }
#pragma unroll
  for (int d = 0; d < 16; ++d) acc[d] = wred_sum(acc[d]);
  if (lane == 0) {
    float* op = O + (((size_t)q * Nb + n) << 6) + h * 16;
#pragma unroll
    for (int d = 0; d < 16; ++d) op[d] = acc[d] * inv;
  }
}

// ---------------- y = LN(x + a) * g + b ; wave per 64-elem row --------------
__global__ void k_add_ln(const float* __restrict__ X, const float* __restrict__ A,
                         const float* __restrict__ g, const float* __restrict__ b,
                         float* __restrict__ Y, int rows) {
  int wid  = (blockIdx.x * blockDim.x + threadIdx.x) >> 5;
  int lane = threadIdx.x & 31;
  if (wid >= rows) return;
  size_t base = (size_t)wid << 6;
  float x0 = X[base + lane]      + A[base + lane];
  float x1 = X[base + 32 + lane] + A[base + 32 + lane];
  float mu = wred_sum(x0 + x1) * (1.f / 64.f);
  float d0 = x0 - mu, d1 = x1 - mu;
  float var = wred_sum(d0 * d0 + d1 * d1) * (1.f / 64.f);
  float rs = rsqrtf(var + EPSF);
  Y[base + lane]      = d0 * rs * g[lane]      + b[lane];
  Y[base + 32 + lane] = d1 * rs * g[lane + 32] + b[lane + 32];
}

// ----------------- final (HW,N,C) -> (N,C,H,W) transpose --------------------
__global__ void k_transpose_out(const float* __restrict__ S, float* __restrict__ O) {
  int id = blockIdx.x * blockDim.x + threadIdx.x;
  if (id >= kN * kC * kHW) return;
  int hw = id & 1023; int c = (id >> 10) & 63; int n = id >> 16;
  O[id] = S[(((size_t)hw * kN + n) << 6) + c];
}

// ===========================================================================
// Host side
// ===========================================================================
struct LayerP {
  const float *Wq, *bq, *Wo, *bo, *W1, *b1, *W2, *b2, *g1, *be1, *g2, *be2;
};

static LayerP layerP(void* const* din, int base, int i) {
  LayerP p;
  p.Wq  = (const float*)din[base + 0]  + (size_t)i * 3 * kD * kD;
  p.bq  = (const float*)din[base + 1]  + (size_t)i * 3 * kD;
  p.Wo  = (const float*)din[base + 2]  + (size_t)i * kD * kD;
  p.bo  = (const float*)din[base + 3]  + (size_t)i * kD;
  p.W1  = (const float*)din[base + 4]  + (size_t)i * kD * kD;
  p.b1  = (const float*)din[base + 5]  + (size_t)i * kD;
  p.W2  = (const float*)din[base + 6]  + (size_t)i * kD * kD;
  p.b2  = (const float*)din[base + 7]  + (size_t)i * kD;
  p.g1  = (const float*)din[base + 8]  + (size_t)i * kD;
  p.be1 = (const float*)din[base + 9]  + (size_t)i * kD;
  p.g2  = (const float*)din[base + 10] + (size_t)i * kD;
  p.be2 = (const float*)din[base + 11] + (size_t)i * kD;
  return p;
}

static void launch_gemm(hipStream_t st, const float* A, const float* W, const float* bias,
                        float* Y, int M, int K, int relu) {
  long long thr = (long long)(M >> 4) * 32; // one wave per 16x64 strip
  k_gemm_xwt64<<<(int)CEILD(thr, 256), 256, 0, st>>>(A, W, bias, Y, M, K, relu);
}

static void run_layer(hipStream_t st, float* state, const float* kin, const float* vin,
                      int Qs, int Ks, const LayerP& p,
                      float* QB, float* KB, float* VB, float* AO, float* OP,
                      float* Q1, float* HB) {
  const int Nb = kN;
  int Mq = Qs * Nb, Mk = Ks * Nb;
  launch_gemm(st, state, p.Wq,             p.bq,       QB, Mq, 64, 0);
  launch_gemm(st, kin,   p.Wq + 64 * 64,   p.bq + 64,  KB, Mk, 64, 0);
  launch_gemm(st, vin,   p.Wq + 128 * 64,  p.bq + 128, VB, Mk, 64, 0);
  {
    long long thr = (long long)Qs * Nb * kNH * 32;
    k_mha<<<(int)CEILD(thr, 256), 256, 0, st>>>(QB, KB, VB, AO, Qs, Ks, Nb);
  }
  launch_gemm(st, AO, p.Wo, p.bo, OP, Mq, 64, 0);
  k_add_ln<<<CEILD(Mq * 32, 256), 256, 0, st>>>(state, OP, p.g1, p.be1, Q1, Mq);
  launch_gemm(st, Q1, p.W1, p.b1, HB, Mq, 64, 1);
  launch_gemm(st, HB, p.W2, p.b2, OP, Mq, 64, 0);
  k_add_ln<<<CEILD(Mq * 32, 256), 256, 0, st>>>(Q1, OP, p.g2, p.be2, state, Mq);
}

extern "C" void kernel_launch(void* const* d_in, const int* in_sizes, int n_in,
                              void* d_out, int out_size, void* d_ws, size_t ws_size,
                              hipStream_t stream) {
  (void)in_sizes; (void)n_in; (void)out_size; (void)ws_size;
  const float* attn_map      = (const float*)d_in[0];
  const float* text_logits   = (const float*)d_in[1];
  const float* image_feature = (const float*)d_in[2];
  const int*   pt_lengths    = (const int*)d_in[3];
  const float* text_proj_W   = (const float*)d_in[4];
  const float* ac1_W  = (const float*)d_in[5];
  const float* bn1g = (const float*)d_in[6],  *bn1b = (const float*)d_in[7];
  const float* bn1m = (const float*)d_in[8],  *bn1v = (const float*)d_in[9];
  const float* pr1  = (const float*)d_in[10];
  const float* ac2_W  = (const float*)d_in[11];
  const float* bn2g = (const float*)d_in[12], *bn2b = (const float*)d_in[13];
  const float* bn2m = (const float*)d_in[14], *bn2v = (const float*)d_in[15];
  const float* pr2  = (const float*)d_in[16];
  const float* ac3_W  = (const float*)d_in[17];
  const float* ac3_b  = (const float*)d_in[18];
  const float* dwc_W  = (const float*)d_in[19];
  const float* dwc_b  = (const float*)d_in[20];

  // --------------- workspace layout (~507 MB of fp32) -----------------------
  float* ws = (float*)d_ws;
  size_t off = 0;
  auto nxt = [&](size_t n) { float* p = ws + off; off += (n + 63) & ~(size_t)63; return p; };
  float* PE    = nxt(1024 * 64);
  float* PET   = nxt(1024 * 64);
  float* POSW  = nxt((size_t)kN * kHW);
  int*   ORDER = (int*)nxt((size_t)kN * kHW);
  float* TF    = nxt((size_t)kT * kN * kD);
  float* TK    = nxt((size_t)kT * kN * kD);
  float* SELF  = nxt((size_t)kSEL * kN * kD);
  float* PEF   = nxt((size_t)kHW * kN * kD);
  float* QB    = nxt((size_t)kHW * kN * kD);
  float* KB    = nxt((size_t)kSEL * kN * kD);
  float* VB    = nxt((size_t)kSEL * kN * kD);
  float* AO    = nxt((size_t)kHW * kN * kD);
  float* OP    = nxt((size_t)kHW * kN * kD);
  float* Q1    = nxt((size_t)kHW * kN * kD);
  float* HB    = nxt((size_t)kHW * kN * kD);
  // conv-phase aliases (convs complete before transformer phase starts)
  float* X1     = AO;  // (N,32,HW)
  float* X2     = QB;  // (N,64,HW)
  float* PW1    = OP;  // (N,64,HW) -> softmaxed in place
  float* CONVIN = Q1;  // instance-norm * pw1, (N,64,HW)

  const int convWaves = (kN * kHW) >> 4; // one wave per 16x64 strip

  // ------------------------- prep phase -------------------------------------
  k_init_pe<<<CEILD(1024 * 32, 256), 256, 0, stream>>>(PE, PET);
  k_posweight<<<CEILD(kN * kHW, 256), 256, 0, stream>>>(attn_map, pt_lengths, POSW);
  k_conv1_bn_prelu<<<CEILD(kN * 32 * kHW, 256), 256, 0, stream>>>(
      POSW, ac1_W, bn1g, bn1b, bn1m, bn1v, pr1, X1);
  k_conv3x3_wmma64<<<CEILD(convWaves * 32, 256), 256, 0, stream>>>(
      X1, ac2_W, nullptr, nullptr, X2, 32, 0);
  k_bn_prelu<<<CEILD(kN * 64 * kHW, 256), 256, 0, stream>>>(
      X2, bn2g, bn2b, bn2m, bn2v, pr2, kN * 64 * kHW, 64);
  k_conv3x3_wmma64<<<CEILD(convWaves * 32, 256), 256, 0, stream>>>(
      X2, ac3_W, ac3_b, nullptr, PW1, 64, 0);
  k_softmax1024<<<CEILD(kN * 64 * 32, 256), 256, 0, stream>>>(PW1, kN * 64);
  k_innorm_mul<<<CEILD(kN * 64 * 32, 256), 256, 0, stream>>>(image_feature, PW1, CONVIN, kN * 64);
  k_conv3x3_wmma64<<<CEILD(convWaves * 32, 256), 256, 0, stream>>>(
      CONVIN, dwc_W, dwc_b, PE, PEF, 64, 1);
  k_argsort1024<<<kN, 1024, 0, stream>>>(POSW, ORDER);
  k_gather_select<<<CEILD(kSEL * kN * kD, 256), 256, 0, stream>>>(PEF, ORDER, SELF);
  // tf = text proj + attn_map x PE einsum (WMMA GEMM, M=N*T, K=1024)
  launch_gemm(stream, attn_map, PET, nullptr, OP, kN * kT, 1024, 0);
  k_textproj<<<CEILD(kT * kN * kD, 256), 256, 0, stream>>>(text_logits, text_proj_W, OP, TF);

  // ------------------------- encoder (2 layers, self-attn) ------------------
  for (int i = 0; i < 2; ++i) {
    LayerP p = layerP(d_in, 21, i);
    run_layer(stream, TF, TF, TF, kT, kT, p, QB, KB, VB, AO, OP, Q1, HB);
  }
  // tk = tf
  k_copy<<<CEILD(kT * kN * kD, 256), 256, 0, stream>>>(TF, TK, kT * kN * kD);

  // ------------------------- text decoder (3 layers) ------------------------
  for (int i = 0; i < 3; ++i) {
    LayerP p = layerP(d_in, 33, i);
    run_layer(stream, TK, SELF, SELF, kT, kSEL, p, QB, KB, VB, AO, OP, Q1, HB);
  }

  // zero unselected pef rows (after gather, before sdec)
  k_zero_unsel<<<CEILD((kHW - kSEL) * kN * kD, 256), 256, 0, stream>>>(PEF, ORDER);

  // ------------------------- spatial decoder (3 layers) ---------------------
  for (int i = 0; i < 3; ++i) {
    LayerP p = layerP(d_in, 45, i);
    run_layer(stream, PEF, TK, TF, kHW, kT, p, QB, KB, VB, AO, OP, Q1, HB);
  }

  // ------------------------- output -----------------------------------------
  k_transpose_out<<<CEILD(kN * kC * kHW, 256), 256, 0, stream>>>(PEF, (float*)d_out);
}
